// Decoder_64871186039165
// MI455X (gfx1250) — compile-verified, matrix operand order checked by reading, not run on hardware
//
#include <hip/hip_runtime.h>

// ---------------------------------------------------------------------------
// CDNA5 (gfx1250) LSTM decoder w/ attention.
// bf16 WMMA (v_wmma_f32_16x16x32_bf16) GEMMs with fused LSTM / tanh epilogues.
// Weights packed once per call into WMMA B-fragment lane layout (one 32B
// contiguous load per lane per 32x16 tile). Each wave register-blocks two
// 16-row M-tiles so every B fragment feeds 2 WMMAs (halves L2 weight traffic).
// Wave32 throughout.
// ---------------------------------------------------------------------------

typedef __attribute__((ext_vector_type(16))) __bf16 v16bf;
typedef __attribute__((ext_vector_type(8)))  float  v8f;

static constexpr int Bn = 128;   // batch
static constexpr int Tn = 128;   // time steps
static constexpr int Sn = 128;   // source length
static constexpr int Hn = 1024;  // hidden
static constexpr int Dn = 512;   // embedding

#define DEV static __device__ __forceinline__

union Frag { uint4 q[2]; v16bf v; };

// A fragment (16x32 bf16, M x K) from a row-major [*, lda] bf16 buffer.
// lane<16 : M=lane,    K = kbase+0..7  and kbase+16..23
// lane>=16: M=lane-16, K = kbase+8..15 and kbase+24..31
DEV v16bf load_a_frag(const __bf16* A, int lda, int mbase, int kbase, int lane) {
  int m = mbase + (lane & 15);
  int k = kbase + ((lane >> 4) << 3);
  const __bf16* p = A + (size_t)m * lda + k;
  Frag f;
  f.q[0] = *(const uint4*)(p);
  f.q[1] = *(const uint4*)(p + 16);
  return f.v;
}

// B fragment: pre-packed tiles, 512 bf16 per tile, lane-contiguous (32B/lane).
DEV v16bf load_b_frag(const __bf16* Bp, size_t tile, int lane) {
  const __bf16* p = Bp + tile * 512 + (size_t)lane * 16;
  Frag f;
  f.q[0] = *(const uint4*)(p);
  f.q[1] = *(const uint4*)(p + 8);
  return f.v;
}

DEV v8f mma_bf16(v16bf a, v16bf b, v8f c) {
  return __builtin_amdgcn_wmma_f32_16x16x32_bf16(
      /*neg_a=*/false, a, /*neg_b=*/false, b,
      /*c_mod=*/(short)0, c, /*reuse_a=*/false, /*reuse_b=*/false);
}

DEV float sigmoid_(float x) { return 1.f / (1.f + __expf(-x)); }
DEV float tanh_(float x)    { float e = __expf(2.f * x); return 1.f - 2.f / (e + 1.f); }

// ---------------------------------------------------------------------------
// Weight packing: W is row-major [N, K] f32 (gates[m,n] = sum_k x[m,k]*W[n,k]).
// Packed tile grid: tile = nb*(K/32) + kb ; within tile, lane-major, e-minor,
// matching the ISA B-matrix (K x N) VGPR layout.
// ---------------------------------------------------------------------------
__global__ void pack_w(const float* __restrict__ W, __bf16* __restrict__ out,
                       int N, int K) {
  size_t i = (size_t)blockIdx.x * 256 + threadIdx.x;
  if (i >= (size_t)N * K) return;
  size_t tile = i >> 9;
  int lane = (int)((i >> 4) & 31);
  int e    = (int)(i & 15);
  int nkb  = K >> 5;
  int nb = (int)(tile / nkb);
  int kb = (int)(tile % nkb);
  int n = (nb << 4) + (lane & 15);
  int k = (kb << 5) + ((lane >> 4) << 3) + ((e < 8) ? e : e + 8);
  out[i] = (__bf16)W[(size_t)n * K + k];
}

// Embedding gather -> bf16 [T*B, D]
__global__ void gather_emb(const int* __restrict__ tokens,
                           const float* __restrict__ emb,
                           __bf16* __restrict__ out) {
  size_t i = (size_t)blockIdx.x * 256 + threadIdx.x;
  if (i >= (size_t)Tn * Bn * Dn) return;
  int    d  = (int)(i & (Dn - 1));
  size_t tb = i >> 9;                 // Dn == 512
  int tok = tokens[tb];
  out[i] = (__bf16)emb[(size_t)tok * Dn + d];
}

// State init: bf16 copies of h0 per layer, f32 c-state copies, zero input-feed.
__global__ void init_state(const float* __restrict__ h0, const float* __restrict__ c0,
                           __bf16* h0b, __bf16* h1b, float* c0s, float* c1s,
                           __bf16* feed0) {
  int i = blockIdx.x * 256 + threadIdx.x;
  if (i >= Bn * Hn) return;
  h0b[i]   = (__bf16)h0[i];
  h1b[i]   = (__bf16)h0[Bn * Hn + i];
  c0s[i]   = c0[i];
  c1s[i]   = c0[Bn * Hn + i];
  feed0[i] = (__bf16)0.f;
}

// Accumulate one K-segment: 2 M-tiles x 4 gates. acc[mt*4 + g].
DEV void seg4x2(v8f acc[8], const __bf16* A, int lda, const __bf16* Bp, int ldb,
                int kb0, int nkb, int nb, int mbase, int lane) {
  for (int kb = 0; kb < nkb; ++kb) {
    v16bf a0 = load_a_frag(A, lda, mbase,      kb << 5, lane);
    v16bf a1 = load_a_frag(A, lda, mbase + 16, kb << 5, lane);
#pragma unroll
    for (int g = 0; g < 4; ++g) {
      size_t tile = (size_t)((g << 6) + nb) * ldb + (kb0 + kb);  // H/16 == 64
      v16bf b = load_b_frag(Bp, tile, lane);
      acc[g]     = mma_bf16(a0, b, acc[g]);
      acc[4 + g] = mma_bf16(a1, b, acc[4 + g]);
    }
  }
}

// ---------------------------------------------------------------------------
// Fused LSTM-cell GEMM: gates = sum_seg A_s @ W_s^T + b ; then i,f,g,o epilogue.
// Grid: 128 blocks (64 n-blocks x 2 M-halves), 64 threads (2 waves).
// Each wave: two 16-row M-tiles, 4 gate 16x16 tiles at the same n-block.
// ---------------------------------------------------------------------------
__global__ __launch_bounds__(64)
void lstm_gate_gemm(const __bf16* A0, int lda0, const __bf16* B0, int ldb0, int kb00, int nkb0,
                    const __bf16* A1, int lda1, const __bf16* B1, int ldb1, int kb01, int nkb1,
                    const __bf16* A2, int lda2, const __bf16* B2, int ldb2, int kb02, int nkb2,
                    const float* __restrict__ b_ih, const float* __restrict__ b_hh,
                    float* __restrict__ c_state, __bf16* __restrict__ h_out) {
  int lane  = threadIdx.x & 31;
  int wave  = threadIdx.x >> 5;                  // 0..1
  int nb    = blockIdx.x >> 1;                   // 0..63
  int mbase = ((blockIdx.x & 1) << 6) + (wave << 5);  // 0,32,64,96

  v8f zero = {0.f, 0.f, 0.f, 0.f, 0.f, 0.f, 0.f, 0.f};
  v8f acc[8] = {zero, zero, zero, zero, zero, zero, zero, zero};

  seg4x2(acc, A0, lda0, B0, ldb0, kb00, nkb0, nb, mbase, lane);
  seg4x2(acc, A1, lda1, B1, ldb1, kb01, nkb1, nb, mbase, lane);
  if (A2) seg4x2(acc, A2, lda2, B2, ldb2, kb02, nkb2, nb, mbase, lane);

  int n    = (nb << 4) + (lane & 15);
  int moff = (lane >> 4) << 3;
  float bi = b_ih[0 * Hn + n] + b_hh[0 * Hn + n];
  float bf = b_ih[1 * Hn + n] + b_hh[1 * Hn + n];
  float bg = b_ih[2 * Hn + n] + b_hh[2 * Hn + n];
  float bo = b_ih[3 * Hn + n] + b_hh[3 * Hn + n];
#pragma unroll
  for (int mt = 0; mt < 2; ++mt) {
#pragma unroll
    for (int j = 0; j < 8; ++j) {
      int m = mbase + (mt << 4) + moff + j;
      size_t idx = (size_t)m * Hn + n;
      float si = sigmoid_(acc[4 * mt + 0][j] + bi);
      float sf = sigmoid_(acc[4 * mt + 1][j] + bf);
      float tg = tanh_(acc[4 * mt + 2][j] + bg);
      float so = sigmoid_(acc[4 * mt + 3][j] + bo);
      float cn = sf * c_state[idx] + si * tg;
      c_state[idx] = cn;
      h_out[idx] = (__bf16)(so * tanh_(cn));
    }
  }
}

// q = h1 @ W_in^T  (M=128, N=1024, K=1024) -> f32
__global__ __launch_bounds__(64)
void gemm_q(const __bf16* __restrict__ A, const __bf16* __restrict__ Bp,
            float* __restrict__ q) {
  int lane  = threadIdx.x & 31;
  int wave  = threadIdx.x >> 5;
  int nb    = blockIdx.x >> 1;
  int mbase = ((blockIdx.x & 1) << 6) + (wave << 5);
  v8f zero = {0.f, 0.f, 0.f, 0.f, 0.f, 0.f, 0.f, 0.f};
  v8f acc0 = zero, acc1 = zero;
  for (int kb = 0; kb < 32; ++kb) {
    v16bf a0 = load_a_frag(A, Hn, mbase,      kb << 5, lane);
    v16bf a1 = load_a_frag(A, Hn, mbase + 16, kb << 5, lane);
    v16bf b  = load_b_frag(Bp, (size_t)nb * 32 + kb, lane);
    acc0 = mma_bf16(a0, b, acc0);
    acc1 = mma_bf16(a1, b, acc1);
  }
  int n    = (nb << 4) + (lane & 15);
  int moff = (lane >> 4) << 3;
#pragma unroll
  for (int j = 0; j < 8; ++j) {
    q[(size_t)(mbase + moff + j) * Hn + n]      = acc0[j];
    q[(size_t)(mbase + 16 + moff + j) * Hn + n] = acc1[j];
  }
}

// out_t = tanh([wctx, h1] @ W_out^T) -> f32 output slice + bf16 input-feed
__global__ __launch_bounds__(64)
void gemm_out(const __bf16* __restrict__ A0, const __bf16* __restrict__ A1,
              const __bf16* __restrict__ Bp, float* __restrict__ out_t,
              __bf16* __restrict__ feed) {
  int lane  = threadIdx.x & 31;
  int wave  = threadIdx.x >> 5;
  int nb    = blockIdx.x >> 1;
  int mbase = ((blockIdx.x & 1) << 6) + (wave << 5);
  v8f zero = {0.f, 0.f, 0.f, 0.f, 0.f, 0.f, 0.f, 0.f};
  v8f acc0 = zero, acc1 = zero;
  for (int kb = 0; kb < 32; ++kb) {              // wctx segment: k 0..1023
    v16bf a0 = load_a_frag(A0, Hn, mbase,      kb << 5, lane);
    v16bf a1 = load_a_frag(A0, Hn, mbase + 16, kb << 5, lane);
    v16bf b  = load_b_frag(Bp, (size_t)nb * 64 + kb, lane);
    acc0 = mma_bf16(a0, b, acc0);
    acc1 = mma_bf16(a1, b, acc1);
  }
  for (int kb = 0; kb < 32; ++kb) {              // h1 segment: k 1024..2047
    v16bf a0 = load_a_frag(A1, Hn, mbase,      kb << 5, lane);
    v16bf a1 = load_a_frag(A1, Hn, mbase + 16, kb << 5, lane);
    v16bf b  = load_b_frag(Bp, (size_t)nb * 64 + 32 + kb, lane);
    acc0 = mma_bf16(a0, b, acc0);
    acc1 = mma_bf16(a1, b, acc1);
  }
  int n    = (nb << 4) + (lane & 15);
  int moff = (lane >> 4) << 3;
#pragma unroll
  for (int j = 0; j < 8; ++j) {
    size_t i0 = (size_t)(mbase + moff + j) * Hn + n;
    size_t i1 = (size_t)(mbase + 16 + moff + j) * Hn + n;
    float v0 = tanh_(acc0[j]);
    float v1 = tanh_(acc1[j]);
    out_t[i0] = v0; feed[i0] = (__bf16)v0;
    out_t[i1] = v1; feed[i1] = (__bf16)v1;
  }
}

// Attention: scores = context . q ; softmax over S ; wctx = attn . context.
// One block per batch row; ~67 MFLOP/step total -> VALU is the right tool.
__global__ __launch_bounds__(256)
void attention(const float* __restrict__ ctx, const float* __restrict__ q,
               __bf16* __restrict__ wctx) {
  __shared__ float sc[Sn];
  __shared__ float qs[Hn];
  int b = blockIdx.x;
  const float* cb = ctx + (size_t)b * Sn * Hn;
  for (int h = threadIdx.x; h < Hn; h += 256) qs[h] = q[(size_t)b * Hn + h];
  __syncthreads();

  int wave = threadIdx.x >> 5, lane = threadIdx.x & 31;
  for (int s = wave * 16; s < wave * 16 + 16; ++s) {
    const float* row = cb + (size_t)s * Hn;
    float p = 0.f;
    for (int h = lane; h < Hn; h += 32) p += row[h] * qs[h];
#pragma unroll
    for (int m = 16; m >= 1; m >>= 1) p += __shfl_xor(p, m, 32);
    if (lane == 0) sc[s] = p;
  }
  __syncthreads();

  if (threadIdx.x == 0) {
    float mx = sc[0];
    for (int s = 1; s < Sn; ++s) mx = fmaxf(mx, sc[s]);
    float sum = 0.f;
    for (int s = 0; s < Sn; ++s) { float e = __expf(sc[s] - mx); sc[s] = e; sum += e; }
    float inv = 1.f / sum;
    for (int s = 0; s < Sn; ++s) sc[s] *= inv;
  }
  __syncthreads();

  for (int h = threadIdx.x; h < Hn; h += 256) {
    float acc = 0.f;
    for (int s = 0; s < Sn; ++s) acc += sc[s] * cb[(size_t)s * Hn + h];
    wctx[(size_t)b * Hn + h] = (__bf16)acc;
  }
}

// ---------------------------------------------------------------------------
extern "C" void kernel_launch(void* const* d_in, const int* in_sizes, int n_in,
                              void* d_out, int out_size, void* d_ws, size_t ws_size,
                              hipStream_t stream) {
  const int*   tokens  = (const int*)d_in[0];
  const float* h0      = (const float*)d_in[1];
  const float* c0      = (const float*)d_in[2];
  const float* context = (const float*)d_in[3];
  const float* emb     = (const float*)d_in[4];
  const float* W_ih0   = (const float*)d_in[5];
  const float* W_hh0   = (const float*)d_in[6];
  const float* b_ih0   = (const float*)d_in[7];
  const float* b_hh0   = (const float*)d_in[8];
  const float* W_ih1   = (const float*)d_in[9];
  const float* W_hh1   = (const float*)d_in[10];
  const float* b_ih1   = (const float*)d_in[11];
  const float* b_hh1   = (const float*)d_in[12];
  const float* W_in    = (const float*)d_in[13];
  const float* W_out   = (const float*)d_in[14];
  float* out = (float*)d_out;

  // workspace bump allocator (all offsets 256B aligned)
  char* base = (char*)d_ws;
  size_t off = 0;
  auto alloc = [&](size_t bytes) -> void* {
    void* p = base + off;
    off = (off + bytes + 255) & ~(size_t)255;
    return p;
  };
  __bf16* pWih0 = (__bf16*)alloc((size_t)4096 * 1536 * 2);
  __bf16* pWhh0 = (__bf16*)alloc((size_t)4096 * 1024 * 2);
  __bf16* pWih1 = (__bf16*)alloc((size_t)4096 * 1024 * 2);
  __bf16* pWhh1 = (__bf16*)alloc((size_t)4096 * 1024 * 2);
  __bf16* pWin  = (__bf16*)alloc((size_t)1024 * 1024 * 2);
  __bf16* pWout = (__bf16*)alloc((size_t)1024 * 2048 * 2);
  __bf16* embs  = (__bf16*)alloc((size_t)Tn * Bn * Dn * 2);
  __bf16* h0b[2] = {(__bf16*)alloc((size_t)Bn * Hn * 2), (__bf16*)alloc((size_t)Bn * Hn * 2)};
  __bf16* h1b[2] = {(__bf16*)alloc((size_t)Bn * Hn * 2), (__bf16*)alloc((size_t)Bn * Hn * 2)};
  __bf16* feed[2] = {(__bf16*)alloc((size_t)Bn * Hn * 2), (__bf16*)alloc((size_t)Bn * Hn * 2)};
  float* c0s  = (float*)alloc((size_t)Bn * Hn * 4);
  float* c1s  = (float*)alloc((size_t)Bn * Hn * 4);
  float* qbuf = (float*)alloc((size_t)Bn * Hn * 4);
  __bf16* wctx = (__bf16*)alloc((size_t)Bn * Hn * 2);

  auto grid_of = [](size_t n) { return (unsigned)((n + 255) / 256); };

  // --- setup: pack weights (L2-resident bf16 B-fragments), gather, init ---
  pack_w<<<grid_of((size_t)4096 * 1536), 256, 0, stream>>>(W_ih0, pWih0, 4096, 1536);
  pack_w<<<grid_of((size_t)4096 * 1024), 256, 0, stream>>>(W_hh0, pWhh0, 4096, 1024);
  pack_w<<<grid_of((size_t)4096 * 1024), 256, 0, stream>>>(W_ih1, pWih1, 4096, 1024);
  pack_w<<<grid_of((size_t)4096 * 1024), 256, 0, stream>>>(W_hh1, pWhh1, 4096, 1024);
  pack_w<<<grid_of((size_t)1024 * 1024), 256, 0, stream>>>(W_in,  pWin,  1024, 1024);
  pack_w<<<grid_of((size_t)1024 * 2048), 256, 0, stream>>>(W_out, pWout, 1024, 2048);
  gather_emb<<<grid_of((size_t)Tn * Bn * Dn), 256, 0, stream>>>(tokens, emb, embs);
  init_state<<<grid_of((size_t)Bn * Hn), 256, 0, stream>>>(h0, c0, h0b[0], h1b[0],
                                                           c0s, c1s, feed[0]);

  // --- recurrent loop: 5 stream-ordered kernels per step ---
  for (int t = 0; t < Tn; ++t) {
    int rp = t & 1, wp = rp ^ 1;

    // Layer 0: gates = emb_t@Wih0[:, :512]^T + feed@Wih0[:, 512:]^T + h0@Whh0^T
    lstm_gate_gemm<<<128, 64, 0, stream>>>(
        embs + (size_t)t * Bn * Dn, Dn, pWih0, 48, 0, 16,
        feed[rp], Hn, pWih0, 48, 16, 32,
        h0b[rp],  Hn, pWhh0, 32, 0, 32,
        b_ih0, b_hh0, c0s, h0b[wp]);

    // Layer 1: gates = h0n@Wih1^T + h1@Whh1^T
    lstm_gate_gemm<<<128, 64, 0, stream>>>(
        h0b[wp], Hn, pWih1, 32, 0, 32,
        h1b[rp], Hn, pWhh1, 32, 0, 32,
        nullptr, 0, nullptr, 0, 0, 0,
        b_ih1, b_hh1, c1s, h1b[wp]);

    gemm_q<<<128, 64, 0, stream>>>(h1b[wp], pWin, qbuf);
    attention<<<128, 256, 0, stream>>>(context, qbuf, wctx);
    gemm_out<<<128, 64, 0, stream>>>(wctx, h1b[wp], pWout,
                                     out + (size_t)t * Bn * Hn, feed[wp]);
  }
}
